// DeepInterestNetworkFixEmb_18786186953288
// MI455X (gfx1250) — matrix operand order repaired and checked by reading, not compiled, not run on hardware
//
#include <hip/hip_runtime.h>

// ---------------------------------------------------------------------------
// DIN (Deep Interest Network) inference for MI455X (gfx1250, wave32, WMMA)
// ---------------------------------------------------------------------------
// B=4096, L=200, D=32.  All GEMMs run as v_wmma_f32_16x16x32_bf16 (fp32 acc).

#define BATCH 4096
#define HISTL 200
#define EDIM  32

#define NEG_INF (-4294967295.0f)

typedef __attribute__((ext_vector_type(16))) __bf16 v16bf;
typedef __attribute__((ext_vector_type(8)))  float  v8f;

union AF { v16bf v; uint32_t u[8]; };

// Workspace layout (bytes, all 256B aligned)
#define WS_W1B 0u        // layer1 attn B-frags: 5 ntiles x 4 ksteps x 512 bf16 = 20480 B
#define WS_W2B 20480u    // layer2 attn B-frags: 3 x 3 x 512 bf16              =  9216 B
#define WS_M1B 29696u    // mlp w1  B-frags: 13 x 9 x 512 bf16                 = 119808 B
#define WS_M2B 149504u   // mlp w2  B-frags: 5 x 7 x 512 bf16                  =  35840 B
#define WS_WQ  185344u   // (Wq + Wdiff) f32 80x64                             =  20480 B
#define WS_X   205824u   // final MLP input, bf16 4096x288                     = 2359296 B

__device__ __forceinline__ int a_kmap(int lane, int e) {
    // 16-bit A-matrix 16x32, contiguous k order: k = e + 16*(lane>=16)
    if (lane < 16) return e;
    else           return e + 16;
}

__device__ __forceinline__ v16bf load_bfrag(const __bf16* buf, int tile, int lane) {
    return *(const v16bf*)(buf + ((tile * 32 + lane) << 4));
}

#define WMMA_BF16(A, Bm, C) \
    __builtin_amdgcn_wmma_f32_16x16x32_bf16(false, (A), false, (Bm), (short)0, (C), false, false)

__device__ __forceinline__ float sigmoidf_(float x) { return 1.0f / (1.0f + __expf(-x)); }

// ---------------------------------------------------------------------------
// Kernel 0: swizzle all weights into WMMA B-fragment order (bf16, padded).
// B-frag element (lane,e): n = ntile*16 + (lane&15), k = ks*32 + e + 16*(lane>=16)
// ---------------------------------------------------------------------------
__global__ __launch_bounds__(256) void prep_kernel(
    const float* __restrict__ att_w1, const float* __restrict__ att_w2,
    const float* __restrict__ mlp_w1, const float* __restrict__ mlp_w2,
    __bf16* __restrict__ W1B, __bf16* __restrict__ W2B,
    __bf16* __restrict__ M1B, __bf16* __restrict__ M2B,
    float* __restrict__ WQ)
{
    int idx = blockIdx.x * 256 + threadIdx.x;

    if (idx < 10240) {  // W1B: folded [Wbeh - Wdiff ; Wprod], K=128, N=80
        int e = idx & 15, lane = (idx >> 4) & 31, tks = idx >> 9;
        int ks = tks & 3, nt = tks >> 2;
        int n = nt * 16 + (lane & 15);
        int k = ks * 32 + e + ((lane & 16) ? 16 : 0);
        float w;
        if (k < 64) w = att_w1[n * 256 + 64 + k] - att_w1[n * 256 + 128 + k];
        else        w = att_w1[n * 256 + 192 + (k - 64)];
        W1B[idx] = (__bf16)w;
        return;
    }
    idx -= 10240;
    if (idx < 4608) {  // W2B: att_w2 (40x80) -> K pad 96, N pad 48
        int e = idx & 15, lane = (idx >> 4) & 31, tks = idx >> 9;
        int ks = tks % 3, nt = tks / 3;
        int n = nt * 16 + (lane & 15);
        int k = ks * 32 + e + ((lane & 16) ? 16 : 0);
        float w = (n < 40 && k < 80) ? att_w2[n * 80 + k] : 0.0f;
        W2B[idx] = (__bf16)w;
        return;
    }
    idx -= 4608;
    if (idx < 59904) {  // M1B: mlp_w1 (200x288) -> K=288, N pad 208
        int e = idx & 15, lane = (idx >> 4) & 31, tks = idx >> 9;
        int ks = tks % 9, nt = tks / 9;
        int n = nt * 16 + (lane & 15);
        int k = ks * 32 + e + ((lane & 16) ? 16 : 0);
        float w = (n < 200) ? mlp_w1[n * 288 + k] : 0.0f;
        M1B[idx] = (__bf16)w;
        return;
    }
    idx -= 59904;
    if (idx < 17920) {  // M2B: mlp_w2 (80x200) -> K pad 224, N=80
        int e = idx & 15, lane = (idx >> 4) & 31, tks = idx >> 9;
        int ks = tks % 7, nt = tks / 7;
        int n = nt * 16 + (lane & 15);
        int k = ks * 32 + e + ((lane & 16) ? 16 : 0);
        float w = (k < 200) ? mlp_w2[n * 200 + k] : 0.0f;
        M2B[idx] = (__bf16)w;
        return;
    }
    idx -= 17920;
    if (idx < 5120) {  // WQ = Wq + Wdiff (f32, 80x64) for the per-batch rank-1 term
        int h = idx >> 6, d2 = idx & 63;
        WQ[idx] = att_w1[h * 256 + d2] + att_w1[h * 256 + 128 + d2];
    }
}

// ---------------------------------------------------------------------------
// Kernel 1: attention + masked softmax + pooling; one workgroup per batch row.
// 256 threads = 8 wave32; each wave owns 16-row M-tiles of the L dimension.
// ---------------------------------------------------------------------------
__global__ __launch_bounds__(256) void attn_kernel(
    const int* __restrict__ uid, const int* __restrict__ gid, const int* __restrict__ cid,
    const int* __restrict__ gid_his, const int* __restrict__ cid_his, const int* __restrict__ masks,
    const float* __restrict__ user_emb, const float* __restrict__ good_emb, const float* __restrict__ cat_emb,
    const float* __restrict__ att_b1, const float* __restrict__ att_b2,
    const float* __restrict__ att_w3, const float* __restrict__ att_b3,
    const __bf16* __restrict__ W1B, const __bf16* __restrict__ W2B, const float* __restrict__ WQ,
    __bf16* __restrict__ Xbf)
{
    __shared__ __bf16 behBf[208 * 64];     // behavior rows (bf16), M padded to 208
    __shared__ __bf16 htile[8][16 * 96];   // per-wave layer-1 activations, K pad 96
    __shared__ float  qv[64];
    __shared__ float  c1v[80];
    __shared__ float  score[208];
    __shared__ float  sred[256];
    __shared__ float  attP[64];
    __shared__ float  ubsP[64];

    const int b    = blockIdx.x;
    const int tid  = threadIdx.x;
    const int wave = tid >> 5;
    const int lane = tid & 31;

    // ---- stage query, init accumulators & scores ----
    if (tid < 64) {
        qv[tid] = (tid < 32) ? good_emb[gid[b] * EDIM + tid]
                             : cat_emb[cid[b] * EDIM + (tid - 32)];
        attP[tid] = 0.0f;
        ubsP[tid] = 0.0f;
    }
    if (tid < 208) score[tid] = att_b3[0];

    // ---- gather behavior history into LDS as bf16 (L2-resident tables) ----
    for (int idx = tid; idx < 208 * 64; idx += 256) {
        int l = idx >> 6, d = idx & 63;
        float v = 0.0f;
        if (l < HISTL) {
            v = (d < 32) ? good_emb[gid_his[b * HISTL + l] * EDIM + d]
                         : cat_emb[cid_his[b * HISTL + l] * EDIM + (d - 32)];
        }
        behBf[idx] = (__bf16)v;
    }
    __syncthreads();

    // ---- per-batch rank-1 term: c1 = b1 + (Wq + Wdiff) * q ----
    if (tid < 80) {
        float s = att_b1[tid];
        for (int d = 0; d < 64; ++d) s += WQ[tid * 64 + d] * qv[d];
        c1v[tid] = s;
    }
    __syncthreads();

    // ---- attention MLP via WMMA, per-wave 16-row tiles ----
    for (int mt = wave; mt < 13; mt += 8) {
        const int row  = lane & 15;
        const int hi8  = (lane & 16) ? 8 : 0;
        const __bf16* arow = &behBf[(mt * 16 + row) * 64];

        AF af[4];
        #pragma unroll
        for (int ks = 0; ks < 2; ++ks)
            #pragma unroll
            for (int p = 0; p < 8; ++p) {
                int k = ks * 32 + a_kmap(lane, 2 * p);
                af[ks].u[p] = *(const uint32_t*)(arow + k);   // beh half (K 0..63)
            }
        #pragma unroll
        for (int ks = 2; ks < 4; ++ks)
            #pragma unroll
            for (int p = 0; p < 8; ++p) {                      // q*beh half (K 64..127)
                int kf = ks * 32 + a_kmap(lane, 2 * p) - 64;
                af[ks].v[2 * p]     = (__bf16)(qv[kf]     * (float)arow[kf]);
                af[ks].v[2 * p + 1] = (__bf16)(qv[kf + 1] * (float)arow[kf + 1]);
            }

        // layer 1: (16x128) x (128x80) + c1, sigmoid -> bf16 per-wave LDS tile
        __bf16* ht = &htile[wave][0];
        for (int nt = 0; nt < 5; ++nt) {
            v8f acc = {};
            #pragma unroll
            for (int ks = 0; ks < 4; ++ks)
                acc = WMMA_BF16(af[ks].v, load_bfrag(W1B, nt * 4 + ks, lane), acc);
            int n = nt * 16 + (lane & 15);
            #pragma unroll
            for (int r = 0; r < 8; ++r)
                ht[(r + hi8) * 96 + n] = (__bf16)sigmoidf_(acc[r] + c1v[n]);
        }
        {   // zero K padding columns 80..95
            int n = 80 + (lane & 15);
            #pragma unroll
            for (int r = 0; r < 8; ++r) ht[(r + hi8) * 96 + n] = (__bf16)0.0f;
        }
        __builtin_amdgcn_wave_barrier();   // order LDS writes before re-reads

        // layer 2: (16x96) x (96x48), sigmoid, fold x att_w3 into per-row score
        for (int nt = 0; nt < 3; ++nt) {
            v8f acc = {};
            #pragma unroll
            for (int ks = 0; ks < 3; ++ks) {
                AF a2f;
                #pragma unroll
                for (int p = 0; p < 8; ++p) {
                    int k = ks * 32 + a_kmap(lane, 2 * p);
                    a2f.u[p] = *(const uint32_t*)(ht + row * 96 + k);
                }
                acc = WMMA_BF16(a2f.v, load_bfrag(W2B, nt * 3 + ks, lane), acc);
            }
            int n = nt * 16 + (lane & 15);
            if (n < 40) {
                float w3 = att_w3[n], bb = att_b2[n];
                #pragma unroll
                for (int r = 0; r < 8; ++r) {
                    int m = mt * 16 + r + hi8;
                    if (m < HISTL)
                        atomicAdd(&score[m], sigmoidf_(acc[r] + bb) * w3);
                }
            }
        }
    }
    __syncthreads();

    // ---- masked softmax over L (LDS tree reductions) ----
    float sval = NEG_INF;
    if (tid < HISTL) sval = (masks[b * HISTL + tid] != 0) ? score[tid] : NEG_INF;
    sred[tid] = sval;
    __syncthreads();
    for (int s = 128; s > 0; s >>= 1) {
        if (tid < s) sred[tid] = fmaxf(sred[tid], sred[tid + s]);
        __syncthreads();
    }
    float smax = sred[0];
    __syncthreads();
    float e = (tid < HISTL) ? __expf(sval - smax) : 0.0f;
    sred[tid] = e;
    __syncthreads();
    for (int s = 128; s > 0; s >>= 1) {
        if (tid < s) sred[tid] += sred[tid + s];
        __syncthreads();
    }
    float inv = 1.0f / sred[0];
    __syncthreads();
    if (tid < 208) score[tid] = (tid < HISTL) ? e * inv : 0.0f;
    __syncthreads();

    // ---- pooling: att = sum w_l * beh_l ; ubs = sum beh_l ----
    {
        int d    = tid & 63;
        int part = tid >> 6;           // 4 partial sums per column
        float sa = 0.0f, su = 0.0f;
        for (int l = part; l < HISTL; l += 4) {
            float bv = (float)behBf[l * 64 + d];
            su += bv;
            sa += score[l] * bv;
        }
        atomicAdd(&attP[d], sa);
        atomicAdd(&ubsP[d], su);
    }
    __syncthreads();

    // ---- emit final-MLP input x = [user, query, ubs, ubs*query, att] (bf16) ----
    if (tid < 288) {
        float v;
        if      (tid < 32)  v = user_emb[uid[b] * EDIM + tid];
        else if (tid < 96)  v = qv[tid - 32];
        else if (tid < 160) v = ubsP[tid - 96];
        else if (tid < 224) v = ubsP[tid - 160] * qv[tid - 160];
        else                v = attP[tid - 224];
        Xbf[b * 288 + tid] = (__bf16)v;
    }
}

// ---------------------------------------------------------------------------
// Kernel 2: final MLP 288 -> 200 -> 80 -> 2 (PReLU), 64 batch rows per block.
// ---------------------------------------------------------------------------
__global__ __launch_bounds__(256) void mlp_kernel(
    const __bf16* __restrict__ Xbf, const __bf16* __restrict__ M1B, const __bf16* __restrict__ M2B,
    const float* __restrict__ b1, const float* __restrict__ b2,
    const float* __restrict__ w3, const float* __restrict__ b3,
    const float* __restrict__ a1, const float* __restrict__ a2,
    float* __restrict__ out)
{
    __shared__ __bf16 H1[64 * 224];   // layer-1 activations, K pad 224
    __shared__ float  out2[64 * 2];

    const int tid  = threadIdx.x;
    const int wave = tid >> 5;
    const int lane = tid & 31;
    const int rowbase = blockIdx.x * 64;

    if (tid < 128) out2[tid] = b3[tid & 1];
    for (int i = tid; i < 64 * 16; i += 256)   // zero pad cols 208..223
        H1[(i >> 4) * 224 + 208 + (i & 15)] = (__bf16)0.0f;
    __syncthreads();

    // layer 1: (64x288) x (288x200pad208) -> PReLU -> bf16 LDS
    for (int t = wave; t < 52; t += 8) {
        int mt = t / 13, nt = t % 13;
        const int row = lane & 15;
        const int hi8 = (lane & 16) ? 8 : 0;
        const __bf16* arow = Xbf + (size_t)(rowbase + mt * 16 + row) * 288;
        v8f acc = {};
        for (int ks = 0; ks < 9; ++ks) {
            AF af;
            #pragma unroll
            for (int p = 0; p < 8; ++p) {
                int k = ks * 32 + a_kmap(lane, 2 * p);
                af.u[p] = *(const uint32_t*)(arow + k);
            }
            acc = WMMA_BF16(af.v, load_bfrag(M1B, nt * 9 + ks, lane), acc);
        }
        int n = nt * 16 + (lane & 15);
        float bb = (n < 200) ? b1[n] : 0.0f;
        float aa = (n < 200) ? a1[n] : 0.0f;
        #pragma unroll
        for (int r = 0; r < 8; ++r) {
            float v = 0.0f;
            if (n < 200) {
                v = acc[r] + bb;
                v = (v >= 0.0f) ? v : aa * v;
            }
            H1[(mt * 16 + r + hi8) * 224 + n] = (__bf16)v;
        }
    }
    __syncthreads();

    // layer 2: (64x224) x (224x80) -> PReLU -> fold x mlp_w3 (80->2) via LDS adds
    for (int t = wave; t < 20; t += 8) {
        int mt = t / 5, nt = t % 5;
        const int row = lane & 15;
        const int hi8 = (lane & 16) ? 8 : 0;
        const __bf16* arow = &H1[(mt * 16 + row) * 224];
        v8f acc = {};
        for (int ks = 0; ks < 7; ++ks) {
            AF af;
            #pragma unroll
            for (int p = 0; p < 8; ++p) {
                int k = ks * 32 + a_kmap(lane, 2 * p);
                af.u[p] = *(const uint32_t*)(arow + k);
            }
            acc = WMMA_BF16(af.v, load_bfrag(M2B, nt * 7 + ks, lane), acc);
        }
        int n = nt * 16 + (lane & 15);
        float bb = b2[n], aa = a2[n];
        float w30 = w3[n], w31 = w3[80 + n];
        #pragma unroll
        for (int r = 0; r < 8; ++r) {
            float v = acc[r] + bb;
            v = (v >= 0.0f) ? v : aa * v;
            int lr = mt * 16 + r + hi8;
            atomicAdd(&out2[lr * 2 + 0], v * w30);
            atomicAdd(&out2[lr * 2 + 1], v * w31);
        }
    }
    __syncthreads();

    if (tid < 128) out[(size_t)(rowbase + (tid >> 1)) * 2 + (tid & 1)] = out2[tid];
}

// ---------------------------------------------------------------------------
extern "C" void kernel_launch(void* const* d_in, const int* in_sizes, int n_in,
                              void* d_out, int out_size, void* d_ws, size_t ws_size,
                              hipStream_t stream) {
    const int*   uid      = (const int*)d_in[0];
    const int*   gid      = (const int*)d_in[1];
    const int*   cid      = (const int*)d_in[2];
    const int*   gid_his  = (const int*)d_in[3];
    const int*   cid_his  = (const int*)d_in[4];
    const int*   masks    = (const int*)d_in[5];
    const float* user_emb = (const float*)d_in[6];
    const float* good_emb = (const float*)d_in[7];
    const float* cat_emb  = (const float*)d_in[8];
    const float* att_w1   = (const float*)d_in[9];
    const float* att_b1   = (const float*)d_in[10];
    const float* att_w2   = (const float*)d_in[11];
    const float* att_b2   = (const float*)d_in[12];
    const float* att_w3   = (const float*)d_in[13];
    const float* att_b3   = (const float*)d_in[14];
    const float* mlp_w1   = (const float*)d_in[15];
    const float* mlp_b1   = (const float*)d_in[16];
    const float* mlp_w2   = (const float*)d_in[17];
    const float* mlp_b2   = (const float*)d_in[18];
    const float* mlp_w3   = (const float*)d_in[19];
    const float* mlp_b3   = (const float*)d_in[20];
    const float* a1       = (const float*)d_in[21];
    const float* a2       = (const float*)d_in[22];

    char* ws = (char*)d_ws;
    __bf16* W1B = (__bf16*)(ws + WS_W1B);
    __bf16* W2B = (__bf16*)(ws + WS_W2B);
    __bf16* M1B = (__bf16*)(ws + WS_M1B);
    __bf16* M2B = (__bf16*)(ws + WS_M2B);
    float*  WQ  = (float*)(ws + WS_WQ);
    __bf16* Xbf = (__bf16*)(ws + WS_X);

    prep_kernel<<<382, 256, 0, stream>>>(att_w1, att_w2, mlp_w1, mlp_w2,
                                         W1B, W2B, M1B, M2B, WQ);

    attn_kernel<<<BATCH, 256, 0, stream>>>(uid, gid, cid, gid_his, cid_his, masks,
                                           user_emb, good_emb, cat_emb,
                                           att_b1, att_b2, att_w3, att_b3,
                                           W1B, W2B, WQ, Xbf);

    mlp_kernel<<<BATCH / 64, 256, 0, stream>>>(Xbf, M1B, M2B,
                                               mlp_b1, mlp_b2, mlp_w3, mlp_b3,
                                               a1, a2, (float*)d_out);
}
